// Self_Attention_3D_41609643163956
// MI455X (gfx1250) — compile-verified, matrix operand order checked by reading, not run on hardware
//
#include <hip/hip_runtime.h>
#include <hip/hip_bf16.h>
#include <math.h>

typedef __attribute__((ext_vector_type(2)))  float  v2f;
typedef __attribute__((ext_vector_type(8)))  float  v8f;
typedef __attribute__((ext_vector_type(16))) __bf16 v16bf;

#define NB   4
#define NC   128
#define NC8  16
#define NN   4096

__device__ __forceinline__ unsigned short f2bf(float f) {
  unsigned int u = __float_as_uint(f);
  u += 0x7fffu + ((u >> 16) & 1u);           // round-to-nearest-even
  return (unsigned short)(u >> 16);
}

// ---------------------------------------------------------------------------
// Kernel 1: 1x1-conv QKV projections.
//   q  [B][16][N]  f32   (row c contiguous in n)  -> WMMA B operand for S^T
//   kT [B][N][16]  f32   (row n contiguous in c)  -> WMMA A operand for S^T
//   vb [B][128][N] bf16  (row c contiguous in n)  -> WMMA B operand for P*V
// ---------------------------------------------------------------------------
__global__ __launch_bounds__(256)
void qkv_kernel(const float* __restrict__ x,
                const float* __restrict__ Wq, const float* __restrict__ bq,
                const float* __restrict__ Wk, const float* __restrict__ bk,
                const float* __restrict__ Wv, const float* __restrict__ bv,
                float* __restrict__ q, float* __restrict__ kT,
                unsigned short* __restrict__ vb) {
  __shared__ float xs[64][132];              // [n][c], padded row (528B, 16B aligned)
  const int t  = threadIdx.x;
  const int b  = blockIdx.y;
  const int n0 = blockIdx.x * 64;
  const int n  = t & 63;
  const int og = t >> 6;                     // 0..3, uniform per wave -> scalar W loads
  // stage x tile transposed: xs[n][c] = x[b][c][n0+n] (coalesced global reads)
  for (int kk = 0; kk < 32; ++kk) {
    int c = og + 4 * kk;
    xs[n][c] = x[((size_t)(b * NC + c) << 12) + n0 + n];
  }
  __syncthreads();
  const float4* xr = (const float4*)&xs[n][0];
  // q and k: 4 output channels each per thread
  for (int r = 0; r < 4; ++r) {
    int o = og * 4 + r;
    float aq = bq[o], ak = bk[o];
    const float* wq = Wq + o * NC;
    const float* wk = Wk + o * NC;
    #pragma unroll 8
    for (int c4 = 0; c4 < 32; ++c4) {
      float4 f = xr[c4];
      aq += wq[4*c4+0]*f.x + wq[4*c4+1]*f.y + wq[4*c4+2]*f.z + wq[4*c4+3]*f.w;
      ak += wk[4*c4+0]*f.x + wk[4*c4+1]*f.y + wk[4*c4+2]*f.z + wk[4*c4+3]*f.w;
    }
    q [(size_t)(b * NC8 + o) * NN + n0 + n]   = aq;
    kT[((size_t)b * NN + n0 + n) * NC8 + o]   = ak;
  }
  // v: 32 output channels per thread, stored bf16
  for (int r = 0; r < 32; ++r) {
    int o = og * 32 + r;
    float av = bv[o];
    const float* wv = Wv + o * NC;
    #pragma unroll 8
    for (int c4 = 0; c4 < 32; ++c4) {
      float4 f = xr[c4];
      av += wv[4*c4+0]*f.x + wv[4*c4+1]*f.y + wv[4*c4+2]*f.z + wv[4*c4+3]*f.w;
    }
    vb[(size_t)(b * NC + o) * NN + n0 + n] = f2bf(av);
  }
}

// ---------------------------------------------------------------------------
// S^T tile: T[j,i] = sum_c kT[j][c] * q[c][i], M=j N=i K=c(16), via 4 chained
// v_wmma_f32_16x16x4_f32.  A half-split K={0,1|2,3}; B mirrors it.
// ---------------------------------------------------------------------------
__device__ __forceinline__ v8f s_tile(const float* __restrict__ qb,
                                      const v2f aS[4], int ii, int lh, int half) {
  v8f t = {0.f, 0.f, 0.f, 0.f, 0.f, 0.f, 0.f, 0.f};
  #pragma unroll
  for (int s = 0; s < 4; ++s) {
    v2f bq;
    const float* p0 = qb + (size_t)(4 * s + 2 * half) * NN + ii + lh;
    bq.x = p0[0];        // K = 4s + 2*half
    bq.y = p0[NN];       // K = 4s + 2*half + 1
    t = __builtin_amdgcn_wmma_f32_16x16x4_f32(false, aS[s], false, bq,
                                              (short)0, t, false, false);
  }
  return t;
}

// ---------------------------------------------------------------------------
// Kernel 2: fused flash attention.  One wave owns 16 map rows j.
//   Sweep 1: row max (fmax only, no exp), shfl-xor fmax merge.
//   Sweep 2: l = sum exp(s-m) (1 exp/elt), shfl-xor add merge.
//   Sweep 3: recompute S^T, write normalized map coalesced, stage P as bf16 in
//            LDS, reload as 16x32 bf16 A, accumulate O^T with bf16 WMMA.
//   S recomputation is WMMA work that co-executes with the exp-bound VALU
//   stream, so trading a sweep for fewer transcendentals wins.
// ---------------------------------------------------------------------------
__global__ __launch_bounds__(128)
void attn_kernel(const float* __restrict__ q, const float* __restrict__ kT,
                 const unsigned short* __restrict__ vb,
                 const float* __restrict__ x, const float* __restrict__ gamma,
                 float* __restrict__ out, float* __restrict__ map) {
  __shared__ __align__(16) unsigned short ldsP[4][16 * 40]; // per-wave 16 rows x 32 bf16, 80B stride
  const int t    = threadIdx.x;
  const int w    = t >> 5;
  const int lane = t & 31;
  const int half = lane >> 4;
  const int lh   = lane & 15;
  const int b    = blockIdx.y;
  const int j0   = blockIdx.x * 64 + w * 16;
  const float* qb = q + (size_t)b * NC8 * NN;

  // A operand (kT rows for this wave's 16 j's) — invariant over the i loop
  v2f aS[4];
  {
    const float* kr = kT + ((size_t)b * NN + j0 + lh) * NC8 + 2 * half;
    #pragma unroll
    for (int s = 0; s < 4; ++s) aS[s] = *(const v2f*)(kr + 4 * s);
  }

  // ---- Sweep 1: row maxima (full-rate fmax only) ----
  float m_run[8];
  #pragma unroll
  for (int r = 0; r < 8; ++r) m_run[r] = -INFINITY;
  for (int ii = 0; ii < NN; ii += 16) {
    v8f tt = s_tile(qb, aS, ii, lh, half);
    #pragma unroll
    for (int r = 0; r < 8; ++r) m_run[r] = fmaxf(m_run[r], tt[r]);
  }
  #pragma unroll
  for (int d = 1; d < 16; d <<= 1) {
    #pragma unroll
    for (int r = 0; r < 8; ++r)
      m_run[r] = fmaxf(m_run[r], __shfl_xor(m_run[r], d, 32));
  }

  // ---- Sweep 2: l = sum exp(s - m)  (one exp per element) ----
  float l_run[8];
  #pragma unroll
  for (int r = 0; r < 8; ++r) l_run[r] = 0.f;
  for (int ii = 0; ii < NN; ii += 16) {
    v8f tt = s_tile(qb, aS, ii, lh, half);
    #pragma unroll
    for (int r = 0; r < 8; ++r) l_run[r] += __expf(tt[r] - m_run[r]);
  }
  #pragma unroll
  for (int d = 1; d < 16; d <<= 1) {
    #pragma unroll
    for (int r = 0; r < 8; ++r)
      l_run[r] += __shfl_xor(l_run[r], d, 32);
  }
  float inv_l[8];
  #pragma unroll
  for (int r = 0; r < 8; ++r) inv_l[r] = 1.0f / l_run[r];

  // ---- Sweep 3: map write + O^T accumulation ----
  const unsigned short* vbb = vb + (size_t)b * NC * NN;
  float* mapb = map + (size_t)b * NN * NN;
  unsigned short* lw = ldsP[w];
  v8f acc[8];
  #pragma unroll
  for (int cb = 0; cb < 8; ++cb) {
    v8f z = {0.f, 0.f, 0.f, 0.f, 0.f, 0.f, 0.f, 0.f};
    acc[cb] = z;
  }
  for (int i0 = 0; i0 < NN; i0 += 32) {
    #pragma unroll
    for (int sub = 0; sub < 2; ++sub) {
      int ii = i0 + sub * 16;
      v8f tt = s_tile(qb, aS, ii, lh, half);
      #pragma unroll
      for (int r = 0; r < 8; ++r) {
        float p  = __expf(tt[r] - m_run[r]) * inv_l[r];
        int row  = r + 8 * half;
        mapb[(size_t)(j0 + row) * NN + ii + lh] = p;      // coalesced 64B/half-wave
        lw[row * 40 + sub * 16 + lh] = f2bf(p);           // stage for A relayout
      }
    }
    // A: bf16 16x32, lane row = lh; lower half K 0-7 & 16-23, upper K 8-15 & 24-31
    union { uint4 u4[2]; v16bf v; } A;
    A.u4[0] = *(const uint4*)&lw[lh * 40 + half * 8];
    A.u4[1] = *(const uint4*)&lw[lh * 40 + 16 + half * 8];
    #pragma unroll
    for (int cb = 0; cb < 8; ++cb) {
      union { unsigned int u[8]; v16bf v; } B;   // B[i,c] = v[c][i], packed i-pairs
      const unsigned short* vr = vbb + (size_t)(cb * 16 + lh) * NN + i0 + 8 * half;
      #pragma unroll
      for (int n2 = 0; n2 < 4; ++n2) {
        B.u[n2]     = *(const unsigned int*)(vr + 2 * n2);        // K = 2n  (+8 upper)
        B.u[4 + n2] = *(const unsigned int*)(vr + 16 + 2 * n2);   // K = 16+2n (+8 upper)
      }
      acc[cb] = __builtin_amdgcn_wmma_f32_16x16x32_bf16(false, A.v, false, B.v,
                                                        (short)0, acc[cb], false, false);
    }
  }

  // ---- epilogue: out[b][c][j] = x + gamma * O^T[j][c] ----
  const float g = gamma[0];
  #pragma unroll
  for (int cb = 0; cb < 8; ++cb) {
    #pragma unroll
    for (int r = 0; r < 8; ++r) {
      size_t idx = (size_t)(b * NC + cb * 16 + lh) * NN + (j0 + r + 8 * half);
      out[idx] = x[idx] + g * acc[cb][r];
    }
  }
}

extern "C" void kernel_launch(void* const* d_in, const int* in_sizes, int n_in,
                              void* d_out, int out_size, void* d_ws, size_t ws_size,
                              hipStream_t stream) {
  (void)in_sizes; (void)n_in; (void)out_size; (void)ws_size;
  const float* x  = (const float*)d_in[0];
  const float* Wq = (const float*)d_in[1];
  const float* bq = (const float*)d_in[2];
  const float* Wk = (const float*)d_in[3];
  const float* bk = (const float*)d_in[4];
  const float* Wv = (const float*)d_in[5];
  const float* bv = (const float*)d_in[6];
  const float* gm = (const float*)d_in[7];

  // workspace: q (1MB) | kT (1MB) | v_bf16 (4MB)
  float*          q   = (float*)d_ws;
  float*          kTp = q + (size_t)NB * NC8 * NN;
  unsigned short* vbf = (unsigned short*)(kTp + (size_t)NB * NN * NC8);

  float* out = (float*)d_out;
  float* map = out + (size_t)NB * NC * NN;   // [B][N][N] after [B][C][D][H][W]

  qkv_kernel<<<dim3(64, NB), 256, 0, stream>>>(x, Wq, bq, Wk, bk, Wv, bv, q, kTp, vbf);
  attn_kernel<<<dim3(64, NB), 128, 0, stream>>>(q, kTp, vbf, x, gm, out, map);
}
